// VAE_51634096833276
// MI455X (gfx1250) — matrix operand order repaired and checked by reading, not compiled, not run on hardware
//
#include <hip/hip_runtime.h>
#include <hip/hip_bf16.h>

// ---------- types for WMMA ----------
typedef __attribute__((ext_vector_type(16))) __bf16 v16bf;
typedef __attribute__((ext_vector_type(8)))  __bf16 v8bf;
typedef __attribute__((ext_vector_type(8)))  float  v8f;

// ---------- bf16 helpers ----------
// Hardware packed convert: dst[15:0]=bf16(lo), dst[31:16]=bf16(hi), RNE.
__device__ __forceinline__ unsigned pk_bf16(float lo, float hi) {
    unsigned r;
    asm("v_cvt_pk_bf16_f32 %0, %1, %2" : "=v"(r) : "v"(lo), "v"(hi));
    return r;
}
__device__ __forceinline__ unsigned short bf16s(float f) {
    return (unsigned short)(pk_bf16(f, f) & 0xFFFFu);
}
__device__ __forceinline__ float bf2f(unsigned short s) {
    union { unsigned u; float f; } t; t.u = ((unsigned)s) << 16; return t.f;
}
__device__ __forceinline__ float elu_f(float v) { return v > 0.f ? v : __expf(v) - 1.f; }

// ---------- CDNA5 async global->LDS (ASYNCcnt-tracked) ----------
__device__ __forceinline__ void async_load_b128(unsigned lds_off, const void* gaddr) {
    asm volatile("global_load_async_to_lds_b128 %0, %1, off"
                 :: "v"(lds_off), "v"(gaddr) : "memory");
}
template<int N>
__device__ __forceinline__ void wait_async_le() {
    asm volatile("s_wait_asynccnt %0" :: "i"(N) : "memory");
}
__device__ __forceinline__ unsigned lds_off_of(const void* p) {
    return (unsigned)(uintptr_t)p;   // low 32 bits of generic LDS pointer = LDS offset
}

// =====================================================================
// WMMA bf16 GEMM:  C[M,N] = epi(A[M,K] @ W^T + bias)
//   A: f32 (A_BF16=0: v_cvt_pk_bf16_f32 staging, reg-double-buffered global
//      prefetch) or bf16 (A_BF16=1: double-buffered async->LDS pipeline).
//   W: zero-padded bf16 [Npad, ldw] -> staging has NO guards.
//   K must be a multiple of 32 (call sites: 27648, 320).
//   Block tile 128x128, 8 waves (2M x 4N), wave tile 64x32 = 4x2 WMMA frags.
//   EPI: 0 = ELU, 1 = tanh.  OUT_BF16: write bf16 (chained) or f32.
// =====================================================================
#define MBLK 128
#define NBLK 128
#define KBLK 32

template<int EPI, bool A_BF16, bool OUT_BF16>
__global__ __launch_bounds__(256)
void gemm_wmma(const void* __restrict__ Ap, int lda,
               const unsigned short* __restrict__ W, int ldw,
               const float* __restrict__ bias,
               void* __restrict__ Cp, int ldc,
               int N, int K)
{
    __shared__ __attribute__((aligned(16))) unsigned short sA[2][MBLK * KBLK]; // 16 KB
    __shared__ __attribute__((aligned(16))) unsigned short sB[2][NBLK * KBLK]; // 16 KB

    const float*          Af  = (const float*)Ap;
    const unsigned short* Abf = (const unsigned short*)Ap;

    const int tid   = threadIdx.x;
    const int lane  = tid & 31;
    const int wave  = tid >> 5;
    const int mBase = blockIdx.y * MBLK;
    const int nBase = blockIdx.x * NBLK;
    const int wm    = (wave & 1) * 64;    // 2 waves across M
    const int wn    = (wave >> 1) * 32;   // 4 waves across N
    const int half  = lane >> 4;
    const int l16   = lane & 15;

    v8f acc[4][2] = {};

    auto asyncA = [&](int p, int k0) {
        #pragma unroll
        for (int i = 0; i < 2; ++i) {
            int id = i * 256 + tid;
            int row = id >> 2, cg = (id & 3) * 8;
            async_load_b128(lds_off_of(&sA[p][row * KBLK + cg]),
                            Abf + (size_t)(mBase + row) * lda + k0 + cg);
        }
    };
    auto asyncB = [&](int p, int k0) {
        #pragma unroll
        for (int i = 0; i < 2; ++i) {
            int id = i * 256 + tid;
            int row = id >> 2, cg = (id & 3) * 8;
            async_load_b128(lds_off_of(&sB[p][row * KBLK + cg]),
                            W + (size_t)(nBase + row) * ldw + k0 + cg);
        }
    };

    float4 ar[4];
    auto loadA = [&](int k0) {
        #pragma unroll
        for (int i = 0; i < 4; ++i) {
            int id = i * 256 + tid;
            int row = id >> 3, cg = (id & 7) * 4;
            ar[i] = *(const float4*)(Af + (size_t)(mBase + row) * lda + k0 + cg);
        }
    };
    auto storeA = [&]() {   // packed hardware cvt + ds_store_b64, into sA[0]
        #pragma unroll
        for (int i = 0; i < 4; ++i) {
            int id = i * 256 + tid;
            int row = id >> 3, cg = (id & 7) * 4;
            uint2 pck;
            pck.x = pk_bf16(ar[i].x, ar[i].y);
            pck.y = pk_bf16(ar[i].z, ar[i].w);
            *(uint2*)&sA[0][row * KBLK + cg] = pck;
        }
    };

    // ---- pipeline prologue ----
    if constexpr (A_BF16) { asyncA(0, 0); } else { loadA(0); }
    asyncB(0, 0);

    const int NT = K / KBLK;
    for (int t = 0; t < NT; ++t) {
        const int  p    = t & 1;
        const bool more = (t + 1 < NT);
        const int  kn   = (t + 1) * KBLK;

        if constexpr (A_BF16) {
            if (more) { asyncA(p ^ 1, kn); asyncB(p ^ 1, kn); }
            // in-order async completion: <=4 outstanding drains tile t exactly
            if (more) wait_async_le<4>(); else wait_async_le<0>();
        } else {
            if (more) asyncB(p ^ 1, kn);
            storeA();
            if (more) wait_async_le<2>(); else wait_async_le<0>();
        }
        __syncthreads();
        if constexpr (!A_BF16) { if (more) loadA(kn); }   // overlaps the WMMAs

        const unsigned short* pA = A_BF16 ? sA[p] : sA[0];
        const unsigned short* pB = sB[p];

        // ---- fragment gathers (ds_read_b128 pairs, ISA 16-bit layouts) ----
        v16bf afrag[4], bfrag[2];
        #pragma unroll
        for (int f = 0; f < 4; ++f) {
            int row = wm + f * 16 + l16;
            v8bf lo = *(const v8bf*)&pA[row * KBLK + half * 8];
            v8bf hi = *(const v8bf*)&pA[row * KBLK + 16 + half * 8];
            #pragma unroll
            for (int e = 0; e < 8; ++e) { afrag[f][e] = lo[e]; afrag[f][8 + e] = hi[e]; }
        }
        #pragma unroll
        for (int f = 0; f < 2; ++f) {
            int coln = wn + f * 16 + l16;
            v8bf lo = *(const v8bf*)&pB[coln * KBLK + half * 8];
            v8bf hi = *(const v8bf*)&pB[coln * KBLK + 16 + half * 8];
            #pragma unroll
            for (int e = 0; e < 8; ++e) { bfrag[f][e] = lo[e]; bfrag[f][8 + e] = hi[e]; }
        }
        #pragma unroll
        for (int i = 0; i < 4; ++i)
            #pragma unroll
            for (int j = 0; j < 2; ++j)
                acc[i][j] = __builtin_amdgcn_wmma_f32_16x16x32_bf16(
                    false, afrag[i], false, bfrag[j], (short)0, acc[i][j], false, false);
        __syncthreads();
    }

    // ---- epilogue: D layout — VGPR r: M = r + 8*half, N = lane%16 ----
    #pragma unroll
    for (int i = 0; i < 4; ++i) {
        #pragma unroll
        for (int j = 0; j < 2; ++j) {
            int col = nBase + wn + j * 16 + l16;
            if (col < N) {
                float bv = bias[col];
                #pragma unroll
                for (int r = 0; r < 8; ++r) {
                    int row = mBase + wm + i * 16 + half * 8 + r;
                    float v = acc[i][j][r] + bv;
                    v = (EPI == 0) ? elu_f(v) : tanhf(v);
                    if constexpr (OUT_BF16)
                        ((unsigned short*)Cp)[(size_t)row * ldc + col] = bf16s(v);
                    else
                        ((float*)Cp)[(size_t)row * ldc + col] = v;
                }
            }
        }
    }
}

// =====================================================================
// Weight pad+convert: f32 [N,K] -> bf16 [Npad, Kpad], zero padding.
// =====================================================================
__global__ __launch_bounds__(256)
void pad_cvt_kernel(const float* __restrict__ src, unsigned* __restrict__ dst,
                    int N, int K, int Npad, int Kpad)
{
    int idx = blockIdx.x * 256 + threadIdx.x;
    int kh = Kpad >> 1;
    if (idx >= Npad * kh) return;
    int n = idx / kh, k = (idx - n * kh) * 2;
    float a = (n < N && k     < K) ? src[(size_t)n * K + k]     : 0.f;
    float b = (n < N && k + 1 < K) ? src[(size_t)n * K + k + 1] : 0.f;
    dst[idx] = pk_bf16(a, b);
}

// Zero pad columns [300,320) of a bf16 [4096,320] activation buffer.
__global__ __launch_bounds__(256)
void zero_pad_kernel(unsigned short* __restrict__ buf)
{
    int idx = blockIdx.x * 256 + threadIdx.x;
    if (idx >= 4096 * 20) return;
    int r = idx / 20, c = 300 + idx % 20;
    buf[r * 320 + c] = 0;
}

// =====================================================================
// Encoder final projection: h2[4096,4] = h1(bf16) @ eW2^T + eb2
// =====================================================================
__global__ __launch_bounds__(256)
void enc2_kernel(const unsigned short* __restrict__ h1, const float* __restrict__ W,
                 const float* __restrict__ b, float* __restrict__ h2)
{
    int row = blockIdx.x * blockDim.x + threadIdx.x;
    if (row >= 4096) return;
    float s0 = b[0], s1 = b[1], s2 = b[2], s3 = b[3];
    const unsigned short* hr = h1 + (size_t)row * 320;
    for (int k = 0; k < 300; ++k) {
        float x = bf2f(hr[k]);
        s0 += x * W[0 * 300 + k];
        s1 += x * W[1 * 300 + k];
        s2 += x * W[2 * 300 + k];
        s3 += x * W[3 * 300 + k];
    }
    h2[row * 4 + 0] = s0; h2[row * 4 + 1] = s1;
    h2[row * 4 + 2] = s2; h2[row * 4 + 3] = s3;
}

// =====================================================================
// BatchNorm (training stats, biased var) + epsilon = exp(logvar/2)*noise
// =====================================================================
__global__ __launch_bounds__(256)
void bn_eps_kernel(const float* __restrict__ h2, const float* __restrict__ g,
                   const float* __restrict__ beta, const float* __restrict__ noise,
                   float* __restrict__ logvar_out, float* __restrict__ eps_out)
{
    __shared__ float s_sum[256 * 4];
    __shared__ float s_sq[256 * 4];
    __shared__ float mu[4], rstd[4];
    int tid = threadIdx.x;

    float sum[4] = {0, 0, 0, 0}, sq[4] = {0, 0, 0, 0};
    for (int r = tid; r < 4096; r += 256) {
        #pragma unroll
        for (int n = 0; n < 4; ++n) {
            float v = h2[r * 4 + n];
            sum[n] += v; sq[n] += v * v;
        }
    }
    #pragma unroll
    for (int n = 0; n < 4; ++n) { s_sum[tid * 4 + n] = sum[n]; s_sq[tid * 4 + n] = sq[n]; }
    __syncthreads();
    for (int off = 128; off > 0; off >>= 1) {
        if (tid < off) {
            #pragma unroll
            for (int n = 0; n < 4; ++n) {
                s_sum[tid * 4 + n] += s_sum[(tid + off) * 4 + n];
                s_sq[tid * 4 + n]  += s_sq[(tid + off) * 4 + n];
            }
        }
        __syncthreads();
    }
    if (tid < 4) {
        float m = s_sum[tid] * (1.f / 4096.f);
        float v = s_sq[tid] * (1.f / 4096.f) - m * m;
        mu[tid] = m;
        rstd[tid] = rsqrtf(v + 1e-5f);
    }
    __syncthreads();
    for (int r = tid; r < 4096; r += 256) {
        #pragma unroll
        for (int n = 0; n < 4; ++n) {
            float lv = (h2[r * 4 + n] - mu[n]) * rstd[n] * g[n] + beta[n];
            logvar_out[r * 4 + n] = lv;
            eps_out[r * 4 + n]    = __expf(lv * 0.5f) * noise[r * 4 + n];
        }
    }
}

// =====================================================================
// Sequential latent scan (NODE=4), one thread per batch row.
// =====================================================================
__global__ __launch_bounds__(256)
void scan_kernel(const float* __restrict__ eps,
                 const float* __restrict__ Bmat, const float* __restrict__ mask,
                 const float* __restrict__ nW1, const float* __restrict__ nb1,
                 const float* __restrict__ nW2, const float* __restrict__ nb2,
                 const float* __restrict__ nW3, const float* __restrict__ nb3,
                 float* __restrict__ latent_out, float* __restrict__ latent_ws,
                 float* __restrict__ align_out)
{
    int row = blockIdx.x * blockDim.x + threadIdx.x;
    if (row >= 4096) return;
    float lat[4] = {0, 0, 0, 0};
    float e[4], al[4];
    #pragma unroll
    for (int n = 0; n < 4; ++n) e[n] = eps[row * 4 + n];

    #pragma unroll
    for (int j = 0; j < 4; ++j) {
        float inp[5];
        #pragma unroll
        for (int i = 0; i < 4; ++i) inp[i] = mask[i * 4 + j] * Bmat[i * 4 + j] * lat[i];
        inp[4] = e[j];
        float a1[4];
        #pragma unroll
        for (int o = 0; o < 4; ++o) {
            float s = nb1[j * 4 + o];
            #pragma unroll
            for (int i = 0; i < 5; ++i) s += nW1[(j * 4 + o) * 5 + i] * inp[i];
            a1[o] = tanhf(s);
        }
        float a2[2];
        #pragma unroll
        for (int o = 0; o < 2; ++o) {
            float s = nb2[j * 2 + o];
            #pragma unroll
            for (int i = 0; i < 4; ++i) s += nW2[(j * 2 + o) * 4 + i] * a1[i];
            a2[o] = tanhf(s);
        }
        float s = nb3[j];
        s += nW3[j * 2 + 0] * a2[0] + nW3[j * 2 + 1] * a2[1];
        al[j] = s;
        lat[j] = tanhf(s);
    }
    #pragma unroll
    for (int n = 0; n < 4; ++n) {
        latent_out[row * 4 + n] = lat[n];
        latent_ws[row * 4 + n]  = lat[n];
        align_out[row * 4 + n]  = al[n];
    }
}

// =====================================================================
// Decoder first layer (K=4): d0 bf16 [4096,320], cols>=300 zeroed.
// =====================================================================
__global__ __launch_bounds__(256)
void dec0_kernel(const float* __restrict__ lat, const float* __restrict__ W,
                 const float* __restrict__ b, unsigned short* __restrict__ out)
{
    int idx = blockIdx.x * blockDim.x + threadIdx.x;
    if (idx >= 4096 * 320) return;
    int m = idx / 320, n = idx - m * 320;
    unsigned short v = 0;
    if (n < 300) {
        float s = b[n];
        #pragma unroll
        for (int k = 0; k < 4; ++k) s += lat[m * 4 + k] * W[n * 4 + k];
        v = bf16s(elu_f(s));
    }
    out[idx] = v;
}

// =====================================================================
// Host-side launch
// =====================================================================
extern "C" void kernel_launch(void* const* d_in, const int* in_sizes, int n_in,
                              void* d_out, int out_size, void* d_ws, size_t ws_size,
                              hipStream_t stream)
{
    const int B = 4096, IMG = 3 * 96 * 96, H = 300;
    const int HPAD = 320, NPAD = 384;

    const float* image = (const float*)d_in[0];
    const float* label = (const float*)d_in[1];
    const float* noise = (const float*)d_in[2];
    const float* mask  = (const float*)d_in[3];
    const float* Bmat  = (const float*)d_in[4];
    const float* eW0   = (const float*)d_in[5];
    const float* eb0   = (const float*)d_in[6];
    const float* eW1   = (const float*)d_in[7];
    const float* eb1   = (const float*)d_in[8];
    const float* eW2   = (const float*)d_in[9];
    const float* eb2   = (const float*)d_in[10];
    const float* bn_g  = (const float*)d_in[11];
    const float* bn_b  = (const float*)d_in[12];
    const float* nW1   = (const float*)d_in[13];
    const float* nb1   = (const float*)d_in[14];
    const float* nW2   = (const float*)d_in[15];
    const float* nb2   = (const float*)d_in[16];
    const float* nW3   = (const float*)d_in[17];
    const float* nb3   = (const float*)d_in[18];
    const float* dW0   = (const float*)d_in[19];
    const float* db0   = (const float*)d_in[20];
    const float* dW1   = (const float*)d_in[21];
    const float* db1   = (const float*)d_in[22];
    const float* dW2   = (const float*)d_in[23];
    const float* db2   = (const float*)d_in[24];

    float* out = (float*)d_out;
    float* out_logvar = out;
    float* out_prior  = out + 16384;
    float* out_latent = out + 2 * 16384;
    float* out_align  = out + 3 * 16384;
    float* out_xhat   = out + 4 * 16384;

    // ---- workspace layout (bytes, 256-aligned) ----
    char* w = (char*)d_ws;
    auto take = [&](size_t bytes) {
        char* p = w;
        w += (bytes + 255) & ~(size_t)255;
        return p;
    };
    unsigned short* w0bf  = (unsigned short*)take((size_t)NPAD * IMG * 2);
    unsigned short* w1bf  = (unsigned short*)take((size_t)NPAD * HPAD * 2);
    unsigned short* dw1bf = (unsigned short*)take((size_t)NPAD * HPAD * 2);
    unsigned short* dw2bf = (unsigned short*)take((size_t)IMG * HPAD * 2);
    unsigned short* h0bf  = (unsigned short*)take((size_t)B * HPAD * 2);
    unsigned short* h1bf  = (unsigned short*)take((size_t)B * HPAD * 2);
    unsigned short* d0bf  = (unsigned short*)take((size_t)B * HPAD * 2);
    unsigned short* d1bf  = (unsigned short*)take((size_t)B * HPAD * 2);
    float* h2        = (float*)take((size_t)B * 4 * 4);
    float* epsb      = (float*)take((size_t)B * 4 * 4);
    float* latent_ws = (float*)take((size_t)B * 4 * 4);

    dim3 blk(256);
    auto gblk = [](long n) { return dim3((unsigned)((n + 255) / 256)); };

    // ---- weight pad+convert to bf16 (zero-padded => guard-free GEMM staging) ----
    pad_cvt_kernel<<<gblk((long)NPAD * (IMG / 2)), blk, 0, stream>>>(eW0, (unsigned*)w0bf, H, IMG, NPAD, IMG);
    pad_cvt_kernel<<<gblk((long)NPAD * (HPAD / 2)), blk, 0, stream>>>(eW1, (unsigned*)w1bf, H, H, NPAD, HPAD);
    pad_cvt_kernel<<<gblk((long)NPAD * (HPAD / 2)), blk, 0, stream>>>(dW1, (unsigned*)dw1bf, H, H, NPAD, HPAD);
    pad_cvt_kernel<<<gblk((long)IMG * (HPAD / 2)), blk, 0, stream>>>(dW2, (unsigned*)dw2bf, IMG, H, IMG, HPAD);
    zero_pad_kernel<<<gblk(4096 * 20), blk, 0, stream>>>(h0bf);
    zero_pad_kernel<<<gblk(4096 * 20), blk, 0, stream>>>(d1bf);

    // Encoder L0: image(f32) [4096,27648] @ eW0^T, ELU -> h0 bf16
    {
        dim3 grid(NPAD / NBLK, B / MBLK);
        gemm_wmma<0, false, true><<<grid, blk, 0, stream>>>(image, IMG, w0bf, IMG, eb0, h0bf, HPAD, H, IMG);
    }
    // Encoder L1: h0 bf16 @ eW1^T, ELU -> h1 bf16
    {
        dim3 grid(NPAD / NBLK, B / MBLK);
        gemm_wmma<0, true, true><<<grid, blk, 0, stream>>>(h0bf, HPAD, w1bf, HPAD, eb1, h1bf, HPAD, H, HPAD);
    }
    // Encoder L2 (N=4)
    enc2_kernel<<<gblk(B), blk, 0, stream>>>(h1bf, eW2, eb2, h2);

    // BN stats + logvar + epsilon
    bn_eps_kernel<<<dim3(1), blk, 0, stream>>>(h2, bn_g, bn_b, noise, out_logvar, epsb);

    // prior_logvar = label
    hipMemcpyAsync(out_prior, label, (size_t)B * 4 * sizeof(float),
                   hipMemcpyDeviceToDevice, stream);

    // Sequential latent scan
    scan_kernel<<<gblk(B), blk, 0, stream>>>(
        epsb, Bmat, mask, nW1, nb1, nW2, nb2, nW3, nb3,
        out_latent, latent_ws, out_align);

    // Decoder L0 (K=4) -> d0 bf16 padded
    dec0_kernel<<<gblk(4096 * 320), blk, 0, stream>>>(latent_ws, dW0, db0, d0bf);

    // Decoder L1: d0 bf16 @ dW1^T, ELU -> d1 bf16
    {
        dim3 grid(NPAD / NBLK, B / MBLK);
        gemm_wmma<0, true, true><<<grid, blk, 0, stream>>>(d0bf, HPAD, dw1bf, HPAD, db1, d1bf, HPAD, H, HPAD);
    }
    // Decoder L2: d1 bf16 @ dW2^T, tanh -> xhat f32
    {
        dim3 grid(IMG / NBLK, B / MBLK);
        gemm_wmma<1, true, false><<<grid, blk, 0, stream>>>(d1bf, HPAD, dw2bf, HPAD, db2, out_xhat, IMG, IMG, HPAD);
    }
}
